// ResidualMamba_17051020165225
// MI455X (gfx1250) — matrix operand order repaired and checked by reading, not compiled, not run on hardware
//
#include <hip/hip_runtime.h>
#include <hip/hip_bf16.h>

typedef __attribute__((ext_vector_type(16))) __bf16 v16bf;
typedef __attribute__((ext_vector_type(8)))  float  v8f;

__device__ __forceinline__ float4 ld4(const float* p) {
    return *reinterpret_cast<const float4*>(p);
}

__device__ __forceinline__ float softplusf(float v) {
    return (v > 20.f) ? v : log1pf(__expf(v));
}

__device__ __forceinline__ float siluf(float v) {
    return v / (1.f + __expf(-v));
}

__device__ __forceinline__ v16bf cvt16(float4 a0, float4 a1, float4 a2, float4 a3) {
    v16bf f;
    f[0]=(__bf16)a0.x; f[1]=(__bf16)a0.y; f[2]=(__bf16)a0.z; f[3]=(__bf16)a0.w;
    f[4]=(__bf16)a1.x; f[5]=(__bf16)a1.y; f[6]=(__bf16)a1.z; f[7]=(__bf16)a1.w;
    f[8]=(__bf16)a2.x; f[9]=(__bf16)a2.y; f[10]=(__bf16)a2.z; f[11]=(__bf16)a2.w;
    f[12]=(__bf16)a3.x; f[13]=(__bf16)a3.y; f[14]=(__bf16)a3.z; f[15]=(__bf16)a3.w;
    return f;
}

// ---------------------------------------------------------------------------
// Branch-free GEMM core, specialized on NS = number of valid 16-wide N
// subtiles (wave-uniform).  No conditionals inside the K-loop, so the NS
// accumulator fragments stay pinned in registers (no phi-copy churn).
// ---------------------------------------------------------------------------
template<int NS>
__device__ __forceinline__ void gemm_core(
    const float* __restrict__ A, int lda,
    const float* __restrict__ W, int ldb,
    float* __restrict__ C, int ldc,
    const float* __restrict__ bias,
    const float* __restrict__ resid, int ldr,
    int K, int act, int lane, int mbase, int nbase)
{
    v8f acc[NS];
    #pragma unroll
    for (int s = 0; s < NS; ++s) acc[s] = v8f{};

    // A-fragment addressing (16-bit A 16x32 layout):
    //  lane l holds row M = l&15; elems 0..7 -> K=klo..klo+7, 8..15 -> K=khi..khi+7
    const int mrow  = mbase + (lane & 15);
    const int klo   = (lane < 16) ? 0 : 8;
    const int khi   = klo + 16;
    // B-fragment addressing (16-bit B 32x16 layout):
    //  lane l holds col N = l&15; elems 0..7 -> K=kboff..kboff+7, 8..15 -> K=kboff+16..kboff+23
    const int ncol  = lane & 15;
    const int kboff = (lane < 16) ? 0 : 8;

    const float* arow = A + (size_t)mrow * lda;
    const float* brow[NS];
    #pragma unroll
    for (int s = 0; s < NS; ++s)
        brow[s] = W + (size_t)(nbase + s * 16 + ncol) * ldb + kboff;

    for (int kb = 0; kb < K; kb += 32) {
        const float* ar = arow + kb;
        __builtin_prefetch(ar + 32, 0, 3);           // near-cache prefetch of next chunk
        v16bf af = cvt16(ld4(ar + klo), ld4(ar + klo + 4),
                         ld4(ar + khi), ld4(ar + khi + 4));
        #pragma unroll
        for (int s = 0; s < NS; ++s) {
            const float* br = brow[s] + kb;
            __builtin_prefetch(br + 32, 0, 3);
            v16bf bf = cvt16(ld4(br), ld4(br + 4), ld4(br + 16), ld4(br + 20));
            acc[s] = __builtin_amdgcn_wmma_f32_16x16x32_bf16(
                false, af, false, bf, (short)0, acc[s], false, false);
        }
    }

    // C/D layout: elem r -> row = r + (lane>=16 ? 8 : 0), col = lane&15
    const int rbase = mbase + ((lane >= 16) ? 8 : 0);
    #pragma unroll
    for (int s = 0; s < NS; ++s) {
        const int col = nbase + s * 16 + ncol;
        #pragma unroll
        for (int r = 0; r < 8; ++r) {
            const int row = rbase + r;
            float v = acc[s][r];
            if (bias)  v += bias[col];
            if (act == 1) v = softplusf(v);
            if (resid) v += resid[(size_t)row * ldr + col];
            C[(size_t)row * ldc + col] = v;
        }
    }
}

// ---------------------------------------------------------------------------
// Generic GEMM: C[M,N] = act(A[M,K] * W[N,K]^T + bias) + resid
// One wave computes a 16(M) x 64(N) strip via v_wmma_f32_16x16x32_bf16.
// Requires M%16==0, N%16==0, K%32==0 (true for all launches here).
// Subtile count resolved once (scalar switch) -> EXEC all-1s at every WMMA.
// ---------------------------------------------------------------------------
__global__ __launch_bounds__(128)
void gemm_wmma_bf16(const float* __restrict__ A, int lda,
                    const float* __restrict__ W, int ldb,
                    float* __restrict__ C, int ldc,
                    const float* __restrict__ bias,
                    const float* __restrict__ resid, int ldr,
                    int M, int N, int K, int act)
{
    const int lane  = threadIdx.x & 31;
    // Force wave index into an SGPR so all tile logic is provably uniform.
    const int wave  = __builtin_amdgcn_readfirstlane(threadIdx.x >> 5);
    const int mbase = blockIdx.x * 16;
    const int nbase = (blockIdx.y * 4 + wave) * 64;
    if (nbase >= N) return;                 // scalar branch: whole wave exits

    int ns = (N - nbase) >> 4;
    if (ns > 4) ns = 4;
    switch (ns) {                           // scalar dispatch
    case 4: gemm_core<4>(A, lda, W, ldb, C, ldc, bias, resid, ldr, K, act, lane, mbase, nbase); break;
    case 3: gemm_core<3>(A, lda, W, ldb, C, ldc, bias, resid, ldr, K, act, lane, mbase, nbase); break;
    case 2: gemm_core<2>(A, lda, W, ldb, C, ldc, bias, resid, ldr, K, act, lane, mbase, nbase); break;
    default: gemm_core<1>(A, lda, W, ldb, C, ldc, bias, resid, ldr, K, act, lane, mbase, nbase); break;
    }
}

// ---------------------------------------------------------------------------
// Causal depthwise conv1d (k=4) + bias + SiLU.  x_in = xz[:, :, 0:2048].
// ---------------------------------------------------------------------------
__global__ __launch_bounds__(256)
void conv_silu_kernel(const float* __restrict__ xz,
                      const float* __restrict__ cw,   // [2048][1][4]
                      const float* __restrict__ cb,   // [2048]
                      float* __restrict__ u,          // [B*L][2048]
                      int Bsz, int L)
{
    const int E = 2048;
    size_t idx = (size_t)blockIdx.x * blockDim.x + threadIdx.x;
    size_t total = (size_t)Bsz * L * E;
    if (idx >= total) return;
    const int e  = (int)(idx % E);
    const size_t bl = idx / E;
    const int l  = (int)(bl % L);
    const size_t row0 = bl - (size_t)l;   // start row of this batch

    float acc = cb[e];
    const float* w = cw + (size_t)e * 4;
    #pragma unroll
    for (int j = 0; j < 4; ++j) {
        const int t = l + j - 3;
        if (t >= 0) acc += xz[(row0 + t) * 4096 + e] * w[j];
    }
    u[bl * 2048 + e] = siluf(acc);
}

// ---------------------------------------------------------------------------
// Selective scan.  One thread per (batch, channel); h[16] lives in VGPRs.
// Reads u, overwrites it in place with (y + u*D) * silu(z).
// ---------------------------------------------------------------------------
__global__ __launch_bounds__(256)
void mamba_scan_kernel(const float* __restrict__ xz,     // [B*L][4096] (z = cols 2048..)
                       float* __restrict__ u,            // [B*L][2048] in/out
                       const float* __restrict__ xdbl,   // [B*L][96]  (B=64..79, C=80..95)
                       const float* __restrict__ delta,  // [B*L][2048]
                       const float* __restrict__ A_log,  // [2048][16]
                       const float* __restrict__ Dp,     // [2048]
                       int Bsz, int L)
{
    const int E = 2048;
    const int chunks = E / 256;
    const int b = blockIdx.x / chunks;
    const int e = (blockIdx.x % chunks) * 256 + threadIdx.x;

    float Arow[16];
    #pragma unroll
    for (int n = 0; n < 16; ++n) Arow[n] = -__expf(A_log[(size_t)e * 16 + n]);
    const float De = Dp[e];

    float h[16];
    #pragma unroll
    for (int n = 0; n < 16; ++n) h[n] = 0.f;

    const size_t rowbase = (size_t)b * L;
    for (int t = 0; t < L; ++t) {
        const size_t r = rowbase + t;
        const float dlt = delta[r * 2048 + e];
        const float ut  = u[r * 2048 + e];
        const float du  = dlt * ut;
        const float* bc = xdbl + r * 96;      // block-uniform address
        float y = 0.f;
        #pragma unroll
        for (int n = 0; n < 16; ++n) {
            const float dA = __expf(dlt * Arow[n]);
            h[n] = dA * h[n] + du * bc[64 + n];
            y += h[n] * bc[80 + n];
        }
        const float z = xz[r * 4096 + 2048 + e];
        u[r * 2048 + e] = (y + ut * De) * siluf(z);
    }
}

// ---------------------------------------------------------------------------
extern "C" void kernel_launch(void* const* d_in, const int* in_sizes, int n_in,
                              void* d_out, int out_size, void* d_ws, size_t ws_size,
                              hipStream_t stream) {
    const float* x          = (const float*)d_in[0];
    const float* in_proj_w  = (const float*)d_in[1];
    const float* conv_w     = (const float*)d_in[2];
    const float* conv_b     = (const float*)d_in[3];
    const float* x_proj_w   = (const float*)d_in[4];
    const float* dt_proj_w  = (const float*)d_in[5];
    const float* dt_proj_b  = (const float*)d_in[6];
    const float* A_log      = (const float*)d_in[7];
    const float* Dvec       = (const float*)d_in[8];
    const float* out_proj_w = (const float*)d_in[9];
    float* out = (float*)d_out;

    const int Bsz = 2, L = 2048, DM = 1024, E = 2048, DTR = 64;
    const int M = Bsz * L;                         // 4096 rows everywhere

    char* ws = (char*)d_ws;
    float* xz    = (float*)(ws);                                        // M * 4096
    float* u     = (float*)(ws + (size_t)M * 4096 * 4);                 // M * 2048
    float* xdbl  = (float*)(ws + (size_t)M * (4096 + 2048) * 4);        // M * 96
    float* delta = (float*)(ws + (size_t)M * (4096 + 2048 + 96) * 4);   // M * 2048

    const dim3 blk(128);

    // 1) xz = x @ in_proj_w^T                       [4096,1024] x [4096,1024]^T
    gemm_wmma_bf16<<<dim3(M / 16, (2 * E + 255) / 256), blk, 0, stream>>>(
        x, DM, in_proj_w, DM, xz, 2 * E, nullptr, nullptr, 0, M, 2 * E, DM, 0);

    // 2) u = silu(depthwise_conv(x_in) + conv_b)
    conv_silu_kernel<<<(M * E) / 256, 256, 0, stream>>>(xz, conv_w, conv_b, u, Bsz, L);

    // 3) x_dbl = u @ x_proj_w^T                     [4096,2048] x [96,2048]^T
    gemm_wmma_bf16<<<dim3(M / 16, (96 + 255) / 256), blk, 0, stream>>>(
        u, E, x_proj_w, E, xdbl, 96, nullptr, nullptr, 0, M, 96, E, 0);

    // 4) delta = softplus(dt @ dt_proj_w^T + b)     dt = x_dbl[:, 0:64] (lda=96)
    gemm_wmma_bf16<<<dim3(M / 16, (E + 255) / 256), blk, 0, stream>>>(
        xdbl, 96, dt_proj_w, DTR, delta, E, dt_proj_b, nullptr, 0, M, E, DTR, 1);

    // 5) selective scan + skip (u*D) + gate silu(z); overwrites u in place
    mamba_scan_kernel<<<Bsz * (E / 256), 256, 0, stream>>>(
        xz, u, xdbl, delta, A_log, Dvec, Bsz, L);

    // 6) out = x + y_gated @ out_proj_w^T           residual fused
    gemm_wmma_bf16<<<dim3(M / 16, (DM + 255) / 256), blk, 0, stream>>>(
        u, E, out_proj_w, E, out, DM, nullptr, x, DM, M, DM, E, 0);
}
